// DelayedMLP_28381143892853
// MI455X (gfx1250) — matrix operand order repaired and checked by reading, not compiled, not run on hardware
//
#include <hip/hip_runtime.h>

// ---------------------------------------------------------------------------
// DelayedMLP for MI455X (gfx1250, wave32, WMMA).
// Key algebraic reduction: per (b,t) exactly one of the two interleaved rows
// equals x, the other is zero; MLP(0) = W2*relu(b1)+b2 is a constant vector.
// So only B*T rows run through the (bf16 WMMA, f32-accum) fused 2-layer MLP.
// ---------------------------------------------------------------------------

typedef __attribute__((ext_vector_type(16))) __bf16 v16bf;
typedef __attribute__((ext_vector_type(8)))  __bf16 v8bf;
typedef __attribute__((ext_vector_type(8)))  float  v8f;

#define B_DIM 256
#define T_DIM 512
#define DIN   256
#define DHID  1024
#define DOUT  256
#define ROWS  (B_DIM * T_DIM)          // 131072 "real" MLP rows

// LDS tile geometry (bf16 elements). Row padding (+8 bf16 = 16B) makes the
// A-fragment ds_load_b128 row stride = 528B / 2064B -> bank stride 4 mod 64,
// i.e. the 16 lanes of a fragment cover all 64 banks exactly once.
#define LDX 264                        // 256 + 8
#define LDH 1032                       // 1024 + 8
#define SM_X_BYTES   (64 * LDX * 2)    // 33792
#define SM_MASK_OFF  SM_X_BYTES        // 64 mask bytes, padded to 128
#define SM_H_OFF     (SM_X_BYTES + 128)
#define SM_BYTES     (SM_H_OFF + 64 * LDH * 2)   // 166016 B < 320KB WGP LDS

// ---------------------------------------------------------------------------
// Repack a row-major f32 weight W[N][K] into bf16 "B-fragment" order:
// frag f = n_tile*(K/32) + kstep holds a 32x16 (KxN) tile, laid out so lane L
// of a wave loads its 8 VGPRs (16 bf16 = 32B) from one contiguous 32B chunk:
//   elem index = f*512 + lane*16 + j*2 + p
//   n = n_tile*16 + (lane&15),  k = kstep*32 + (j>>2)*16 + (lane>>4)*8 + 2(j&3) + p
// ---------------------------------------------------------------------------
__global__ void pack_w_kernel(const float* __restrict__ W, __bf16* __restrict__ Wp,
                              int N, int K) {
  int tid = blockIdx.x * 256 + threadIdx.x;
  if (tid >= N * K) return;
  int p    = tid & 1;
  int j    = (tid >> 1) & 7;
  int lane = (tid >> 4) & 31;
  int f    = tid >> 9;
  int ksteps = K >> 5;
  int nt = f / ksteps;
  int ks = f - nt * ksteps;
  int n = nt * 16 + (lane & 15);
  int k = ks * 32 + (j >> 2) * 16 + (lane >> 4) * 8 + 2 * (j & 3) + p;
  Wp[tid] = (__bf16)W[(size_t)n * K + k];
}

// Constant output of the all-zero rows: c = W2 * relu(b1) + b2   (f32 exact)
__global__ void zero_row_kernel(const float* __restrict__ W2, const float* __restrict__ b1,
                                const float* __restrict__ b2, float* __restrict__ cst) {
  int o = threadIdx.x;                 // 256 threads
  const float* w = W2 + (size_t)o * DHID;
  float s = b2[o];
  for (int h = 0; h < DHID; ++h) s += fmaxf(b1[h], 0.f) * w[h];
  cst[o] = s;
}

// Gate mask per row, in f32 (sign decision at sigmoid=0.5 <=> logit>=0).
// One wave per row: coalesced 128B loads, shuffle reduction.
__global__ void gate_kernel(const float* __restrict__ x, const float* __restrict__ Wg,
                            const float* __restrict__ bg, unsigned char* __restrict__ msk) {
  int lane = threadIdx.x & 31;
  int wave = threadIdx.x >> 5;
  int r = blockIdx.x * 8 + wave;
  const float* xr = x + (size_t)r * DIN;
  float s = 0.f;
#pragma unroll
  for (int i = 0; i < 8; ++i) s += xr[lane + 32 * i] * Wg[lane + 32 * i];
#pragma unroll
  for (int off = 16; off; off >>= 1) s += __shfl_xor(s, off, 32);
  if (lane == 0) msk[r] = (s + bg[0] >= 0.f) ? 1 : 0;
}

// ---------------------------------------------------------------------------
// Fused 2-layer MLP. Block = 64 rows, 8 waves. Stage 1 keeps the entire
// 64x1024 bf16 hidden tile in LDS (only possible with CDNA5's 320KB WGP LDS).
// Each wave runs two N-tiles per K-loop (two accumulators share one A frag)
// to break the WMMA D->C dependency chain.
// ---------------------------------------------------------------------------
__global__ __launch_bounds__(256, 1)
void fused_mlp_kernel(const float* __restrict__ x,
                      const __bf16* __restrict__ W1p,
                      const __bf16* __restrict__ W2p,
                      const float* __restrict__ b1,
                      const float* __restrict__ b2,
                      const unsigned char* __restrict__ msk,
                      const float* __restrict__ cst,
                      float* __restrict__ out) {
  extern __shared__ char smem[];
  __bf16* Xs = (__bf16*)smem;
  unsigned char* maskS = (unsigned char*)(smem + SM_MASK_OFF);
  __bf16* Hs = (__bf16*)(smem + SM_H_OFF);

  const int tid  = threadIdx.x;
  const int lane = tid & 31;
  const int wave = tid >> 5;
  const int rowBase = blockIdx.x * 64;

  // ---- load + convert X tile: 64 x 256 f32 -> bf16 (coalesced float2) ----
#pragma unroll
  for (int i = 0; i < 32; ++i) {
    int idx = tid + i * 256;           // float2 index 0..8191
    int r   = idx >> 7;
    int c2  = idx & 127;
    float2 v = *(const float2*)(x + (((size_t)(rowBase + r)) << 8) + (c2 << 1));
    union { __bf16 h[2]; unsigned int u; } pk;
    pk.h[0] = (__bf16)v.x;
    pk.h[1] = (__bf16)v.y;
    *(unsigned int*)(Xs + r * LDX + (c2 << 1)) = pk.u;
  }
  if (tid < 64) maskS[tid] = msk[rowBase + tid];
  __syncthreads();

  const int mQuad = (wave & 3) * 16;   // wave's M tile (4 M-tiles x 2 waves)
  const int half  = wave >> 2;         // wave's N half
  const int khalf = lane >> 4;         // A/B fragment K-half select
  const int nlane = lane & 15;
  const int mbase = mQuad + (khalf << 3);  // D-fragment row base for this lane

  // ---- stage 1: H = relu(X @ W1^T + b1), 64x1024 bf16 into LDS ----
  {
    const __bf16* aRow = Xs + (mQuad + nlane) * LDX + (khalf << 3);
    for (int np = 0; np < 16; ++np) {
      int nt = half * 32 + np * 2;
      const __bf16* bp0 = W1p + ((size_t)nt * 8) * 512 + lane * 16;
      const __bf16* bp1 = bp0 + 8 * 512;
      v8f acc0 = {}; v8f acc1 = {};
#pragma unroll
      for (int ks = 0; ks < 8; ++ks) {                 // K = 256 = 8 x 32
        v8bf alo = *(const v8bf*)(aRow + ks * 32);
        v8bf ahi = *(const v8bf*)(aRow + ks * 32 + 16);
        v16bf a = __builtin_shufflevector(alo, ahi, 0,1,2,3,4,5,6,7,
                                          8,9,10,11,12,13,14,15);
        v16bf bb0 = *(const v16bf*)(bp0 + ks * 512);
        v16bf bb1 = *(const v16bf*)(bp1 + ks * 512);
        acc0 = __builtin_amdgcn_wmma_f32_16x16x32_bf16(false, a, false, bb0,
                                                       (short)0, acc0, false, false);
        acc1 = __builtin_amdgcn_wmma_f32_16x16x32_bf16(false, a, false, bb1,
                                                       (short)0, acc1, false, false);
      }
      int nc0 = nt * 16 + nlane;
      int nc1 = nc0 + 16;
      float bia0 = b1[nc0], bia1 = b1[nc1];
#pragma unroll
      for (int i = 0; i < 8; ++i) {
        Hs[(mbase + i) * LDH + nc0] = (__bf16)fmaxf(acc0[i] + bia0, 0.f);
        Hs[(mbase + i) * LDH + nc1] = (__bf16)fmaxf(acc1[i] + bia1, 0.f);
      }
    }
  }
  __syncthreads();

  // ---- stage 2: out = H @ W2^T + b2, scatter by gate mask ----
  {
    const __bf16* aRow = Hs + (mQuad + nlane) * LDH + (khalf << 3);
    for (int np = 0; np < 4; ++np) {
      int nt = half * 8 + np * 2;
      const __bf16* bp0 = W2p + ((size_t)nt * 32) * 512 + lane * 16;
      const __bf16* bp1 = bp0 + 32 * 512;
      v8f acc0 = {}; v8f acc1 = {};
#pragma unroll 8
      for (int ks = 0; ks < 32; ++ks) {                // K = 1024 = 32 x 32
        v8bf alo = *(const v8bf*)(aRow + ks * 32);
        v8bf ahi = *(const v8bf*)(aRow + ks * 32 + 16);
        v16bf a = __builtin_shufflevector(alo, ahi, 0,1,2,3,4,5,6,7,
                                          8,9,10,11,12,13,14,15);
        v16bf bb0 = *(const v16bf*)(bp0 + ks * 512);
        v16bf bb1 = *(const v16bf*)(bp1 + ks * 512);
        acc0 = __builtin_amdgcn_wmma_f32_16x16x32_bf16(false, a, false, bb0,
                                                       (short)0, acc0, false, false);
        acc1 = __builtin_amdgcn_wmma_f32_16x16x32_bf16(false, a, false, bb1,
                                                       (short)0, acc1, false, false);
      }
      int nc0 = nt * 16 + nlane;
      int nc1 = nc0 + 16;
      float bo0 = b2[nc0], bo1 = b2[nc1];
      float cz0 = cst[nc0], cz1 = cst[nc1];
#pragma unroll
      for (int i = 0; i < 8; ++i) {
        int mloc = mbase + i;
        int r    = rowBase + mloc;
        int bidx = r >> 9;                 // / T_DIM
        int tt   = r & (T_DIM - 1);
        int slot = maskS[mloc] ? 0 : 1;    // mask=1 -> x in immediate slot
        size_t obase = ((size_t)bidx * (2 * T_DIM) + 2 * tt) << 8;   // *DOUT
        float* oX = out + obase + ((size_t)slot << 8);
        float* oZ = out + obase + ((size_t)(1 - slot) << 8);
        oX[nc0] = acc0[i] + bo0;
        oX[nc1] = acc1[i] + bo1;
        oZ[nc0] = cz0;
        oZ[nc1] = cz1;
      }
    }
  }
}

// ---------------------------------------------------------------------------
extern "C" void kernel_launch(void* const* d_in, const int* in_sizes, int n_in,
                              void* d_out, int out_size, void* d_ws, size_t ws_size,
                              hipStream_t stream) {
  (void)in_sizes; (void)n_in; (void)out_size; (void)ws_size;
  const float* x  = (const float*)d_in[0];
  const float* W1 = (const float*)d_in[1];
  const float* b1 = (const float*)d_in[2];
  const float* W2 = (const float*)d_in[3];
  const float* b2 = (const float*)d_in[4];
  const float* Wg = (const float*)d_in[5];
  const float* bg = (const float*)d_in[6];
  float* out = (float*)d_out;

  char* ws = (char*)d_ws;
  __bf16* W1p = (__bf16*)ws;                                   // 512 KB
  __bf16* W2p = (__bf16*)(ws + (size_t)512 * 1024);            // 512 KB
  float*  cst = (float*)(ws + (size_t)1024 * 1024);            // 1 KB
  unsigned char* msk = (unsigned char*)(ws + (size_t)1024 * 1024 + 4096); // 128 KB

  pack_w_kernel<<<(DHID * DIN) / 256, 256, 0, stream>>>(W1, W1p, DHID, DIN);
  pack_w_kernel<<<(DOUT * DHID) / 256, 256, 0, stream>>>(W2, W2p, DOUT, DHID);
  zero_row_kernel<<<1, 256, 0, stream>>>(W2, b1, b2, cst);
  gate_kernel<<<ROWS / 8, 256, 0, stream>>>(x, Wg, bg, msk);
  fused_mlp_kernel<<<ROWS / 64, 256, SM_BYTES, stream>>>(x, W1p, W2p, b1, b2,
                                                         msk, cst, out);
}
